// GCN_21887153340779
// MI455X (gfx1250) — matrix operand order repaired and checked by reading, not compiled, not run on hardware
//
#include <hip/hip_runtime.h>
#include <hip/hip_fp16.h>

typedef __attribute__((ext_vector_type(16))) _Float16 v16h;
typedef __attribute__((ext_vector_type(8)))  float    v8f;

#define WAVES_PER_BLOCK 8

// -------------------------------------------------------------------------
// C[M,Nc] = A[M,K] @ W[K,Nc] with W pre-transposed: Wt[Nc,K].
// fp32 in/out via split-f16 WMMA: x = hi + lo; x*w ~= hi*hi + hi*lo + lo*hi
// accumulated in the f32 WMMA accumulator -> ~fp32 accuracy on matrix pipes.
//
// One wave computes a 16x64 tile (4 N-subtiles): the A fragment is loaded and
// split ONCE per K-step and reused for 4 B subtiles / 4 accumulators, cutting
// re-reads of the large A matrix 4x (A is the only big operand; Wt <= 512KB
// stays L2/WGP$-resident). Per K-step: 4+16 b128 loads, 12 WMMAs.
//
// M must be a multiple of 16 (50000 = 3125*16). TAIL=false: Nc % 64 == 0.
// 32-bit indexing throughout (max index ~25.6M) -> no v_mul_u64 in the loop.
// -------------------------------------------------------------------------
template <bool TAIL>
__global__ void __launch_bounds__(32 * WAVES_PER_BLOCK)
gemm_wmma_f32(const float* __restrict__ A, const float* __restrict__ Wt,
              float* __restrict__ C, int K, int Nc, int tilesNG, int nTiles) {
  int wave = threadIdx.x >> 5;
  int tile = blockIdx.x * WAVES_PER_BLOCK + wave;
  if (tile >= nTiles) return;                 // wave-uniform: EXEC stays all-1s
  int tm   = tile / tilesNG;                  // M-tile (16 rows)
  int tg   = tile - tm * tilesNG;             // N-group (64 cols = 4 subtiles)
  int lane = threadIdx.x & 31;
  int half = lane >> 4;                       // 0: lanes 0-15, 1: lanes 16-31
  int l16  = lane & 15;
  int row  = tm * 16 + l16;                   // A-matrix row for this lane

  // Per-subtile column, clamped + 0/1 mask for the TAIL case (unconditional
  // loads -> no EXEC-masked load sequences; WMMA always sees full EXEC).
  int   coln[4], colc[4];
  float cmask[4];
#pragma unroll
  for (int t = 0; t < 4; ++t) {
    coln[t]  = tg * 64 + t * 16 + l16;
    colc[t]  = coln[t];
    cmask[t] = 1.0f;
    if (TAIL) {
      colc[t]  = (coln[t] < Nc) ? coln[t] : (Nc - 1);
      cmask[t] = (coln[t] < Nc) ? 1.0f : 0.0f;
    }
  }

  const float* Arow = A + row * K;
  const float* Brow[4];
#pragma unroll
  for (int t = 0; t < 4; ++t) Brow[t] = Wt + colc[t] * K;

  v8f acc[4] = {{}, {}, {}, {}};
  for (int kk = 0; kk < K; kk += 32) {
    // ---- A fragment (16-bit A 16x32 layout, ISA 7.12.2), loaded once ----
    // lane<16: elems 0-7 -> K=kk+0..7,  elems 8-15 -> K=kk+16..23
    // lane>=16: elems 0-7 -> K=kk+8..15, elems 8-15 -> K=kk+24..31
    int akbase = kk + half * 8;
    float4 a0 = *(const float4*)(Arow + akbase);
    float4 a1 = *(const float4*)(Arow + akbase + 4);
    float4 a2 = *(const float4*)(Arow + akbase + 16);
    float4 a3 = *(const float4*)(Arow + akbase + 20);
    float af[16] = {a0.x, a0.y, a0.z, a0.w, a1.x, a1.y, a1.z, a1.w,
                    a2.x, a2.y, a2.z, a2.w, a3.x, a3.y, a3.z, a3.w};
    v16h ahi, alo;
#pragma unroll
    for (int j = 0; j < 16; ++j) {
      float x = af[j];
      _Float16 h = (_Float16)x;
      ahi[j] = h;
      alo[j] = (_Float16)(x - (float)h);
    }
    // ---- 4 B subtiles (32x16 each), contiguous in K thanks to Wt[Nc,K] ----
    int bkbase = kk + half * 8;
#pragma unroll
    for (int t = 0; t < 4; ++t) {
      float4 b0 = *(const float4*)(Brow[t] + bkbase);
      float4 b1 = *(const float4*)(Brow[t] + bkbase + 4);
      float4 b2 = *(const float4*)(Brow[t] + bkbase + 16);
      float4 b3 = *(const float4*)(Brow[t] + bkbase + 20);
      float bf[16] = {b0.x, b0.y, b0.z, b0.w, b1.x, b1.y, b1.z, b1.w,
                      b2.x, b2.y, b2.z, b2.w, b3.x, b3.y, b3.z, b3.w};
      v16h bhi, blo;
#pragma unroll
      for (int j = 0; j < 16; ++j) {
        float x = bf[j];
        if (TAIL) x *= cmask[t];
        _Float16 h = (_Float16)x;
        bhi[j] = h;
        blo[j] = (_Float16)(x - (float)h);
      }
      // 3-product split accumulation (lo*lo term ~2^-22, dropped)
      acc[t] = __builtin_amdgcn_wmma_f32_16x16x32_f16(false, alo, false, bhi, (short)0, acc[t], false, false);
      acc[t] = __builtin_amdgcn_wmma_f32_16x16x32_f16(false, ahi, false, blo, (short)0, acc[t], false, false);
      acc[t] = __builtin_amdgcn_wmma_f32_16x16x32_f16(false, ahi, false, bhi, (short)0, acc[t], false, false);
    }
  }
  // C/D layout: VGPR r -> row tm*16 + r + 8*half, col = coln[t]
  int rbase = (tm * 16 + 8 * half) * Nc;
#pragma unroll
  for (int t = 0; t < 4; ++t) {
    if (!TAIL || coln[t] < Nc) {
#pragma unroll
      for (int r = 0; r < 8; ++r)
        C[rbase + r * Nc + coln[t]] = acc[t][r];
    }
  }
}

// -------------------------------------------------------------------------
// Wt[Nc,K] = W[K,Nc]^T. Weight matrices are tiny (<=512KB); one-shot cost.
// -------------------------------------------------------------------------
__global__ void transpose_mat(const float* __restrict__ Wsrc, float* __restrict__ Wt,
                              int K, int Nc) {
  int idx = blockIdx.x * blockDim.x + threadIdx.x;
  if (idx >= K * Nc) return;
  int k = idx / Nc;
  int n = idx - k * Nc;
  Wt[n * K + k] = Wsrc[idx];
}

// -------------------------------------------------------------------------
// Build CSR row pointers from sorted COO rows (deterministic, no atomics).
// -------------------------------------------------------------------------
__global__ void build_row_ptr(const int* __restrict__ rows, int E, int N,
                              int* __restrict__ rp) {
  int e = blockIdx.x * blockDim.x + threadIdx.x;
  if (e >= E) return;
  int r    = rows[e];
  int prev = (e == 0) ? -1 : rows[e - 1];
  for (int x = prev + 1; x <= r; ++x) rp[x] = e;
  if (e == E - 1)
    for (int x = r + 1; x <= N; ++x) rp[x] = E;
}

// -------------------------------------------------------------------------
// out[node] = sum_{e in row(node)} vals[e] * S[cols[e]] + bias  (opt. ReLU)
// One block per node; each thread owns 4 contiguous features (float4).
// Gathers are coalesced across the block and L2-resident (S fits in 192MB L2).
// -------------------------------------------------------------------------
__global__ void spmm_bias_act(const int* __restrict__ rp, const int* __restrict__ cols,
                              const float* __restrict__ vals, const float* __restrict__ S,
                              const float* __restrict__ bias, float* __restrict__ out,
                              int d, int relu) {
  int node = blockIdx.x;
  int f4   = threadIdx.x;
  int d4   = d >> 2;
  if (f4 >= d4) return;
  const float4* S4 = (const float4*)S;
  float4 acc = {0.f, 0.f, 0.f, 0.f};
  int e0 = rp[node], e1 = rp[node + 1];
  for (int e = e0; e < e1; ++e) {
    float  w  = vals[e];
    float4 sv = S4[cols[e] * d4 + f4];
    acc.x = fmaf(w, sv.x, acc.x);
    acc.y = fmaf(w, sv.y, acc.y);
    acc.z = fmaf(w, sv.z, acc.z);
    acc.w = fmaf(w, sv.w, acc.w);
  }
  const float4 b = ((const float4*)bias)[f4];
  acc.x += b.x; acc.y += b.y; acc.z += b.z; acc.w += b.w;
  if (relu) {
    acc.x = fmaxf(acc.x, 0.f); acc.y = fmaxf(acc.y, 0.f);
    acc.z = fmaxf(acc.z, 0.f); acc.w = fmaxf(acc.w, 0.f);
  }
  ((float4*)out)[node * d4 + f4] = acc;
}

// -------------------------------------------------------------------------
// In-place row-wise log_softmax, one wave32 per row (d <= 64).
// -------------------------------------------------------------------------
__global__ void log_softmax_rows(float* __restrict__ out, int N, int d) {
  int row = blockIdx.x * (blockDim.x >> 5) + (threadIdx.x >> 5);
  if (row >= N) return;
  int   lane = threadIdx.x & 31;
  float NEG  = -3.402823e38f;
  float v0 = (lane < d)      ? out[row * d + lane]      : NEG;
  float v1 = (lane + 32 < d) ? out[row * d + lane + 32] : NEG;
  float m = fmaxf(v0, v1);
#pragma unroll
  for (int off = 16; off > 0; off >>= 1)
    m = fmaxf(m, __shfl_xor(m, off, 32));
  float s = 0.f;
  if (lane < d)      s += expf(v0 - m);
  if (lane + 32 < d) s += expf(v1 - m);
#pragma unroll
  for (int off = 16; off > 0; off >>= 1)
    s += __shfl_xor(s, off, 32);
  float ls = logf(s) + m;
  if (lane < d)      out[row * d + lane]      = v0 - ls;
  if (lane + 32 < d) out[row * d + lane + 32] = v1 - ls;
}

// -------------------------------------------------------------------------
static inline size_t align256(size_t x) { return (x + 255) & ~(size_t)255; }

static void launch_gemm(const float* A, const float* Wt, float* C,
                        int M, int K, int Nc, hipStream_t stream) {
  int tilesM  = M / 16;
  int tilesNG = (Nc + 63) / 64;               // 64-column groups per wave
  int nTiles  = tilesM * tilesNG;
  int blocks  = (nTiles + WAVES_PER_BLOCK - 1) / WAVES_PER_BLOCK;
  if ((Nc & 63) == 0)
    gemm_wmma_f32<false><<<blocks, 32 * WAVES_PER_BLOCK, 0, stream>>>(A, Wt, C, K, Nc, tilesNG, nTiles);
  else
    gemm_wmma_f32<true><<<blocks, 32 * WAVES_PER_BLOCK, 0, stream>>>(A, Wt, C, K, Nc, tilesNG, nTiles);
}

extern "C" void kernel_launch(void* const* d_in, const int* in_sizes, int n_in,
                              void* d_out, int out_size, void* d_ws, size_t ws_size,
                              hipStream_t stream) {
  const float* x     = (const float*)d_in[0];
  const int*   erow  = (const int*)  d_in[1];
  const int*   ecol  = (const int*)  d_in[2];
  const float* eval_ = (const float*)d_in[3];
  const float* W1    = (const float*)d_in[4];
  const float* b1    = (const float*)d_in[5];
  const float* Wx    = (const float*)d_in[6];
  const float* bx    = (const float*)d_in[7];
  const float* W2    = (const float*)d_in[8];
  const float* b2    = (const float*)d_in[9];

  const int E      = in_sizes[1];
  const int nhid   = in_sizes[5];              // 256
  const int nfeat  = in_sizes[4] / nhid;       // 512
  const int N      = in_sizes[0] / nfeat;      // 50000
  const int nclass = in_sizes[9];              // 40

  // Workspace: row_ptr | support | hA | hB | w1T | wxT | w2T
  char*  ws      = (char*)d_ws;
  int*   rp      = (int*)ws;
  size_t off     = align256((size_t)(N + 1) * sizeof(int));
  float* support = (float*)(ws + off); off += align256((size_t)N * nhid * sizeof(float));
  float* hA      = (float*)(ws + off); off += align256((size_t)N * nhid * sizeof(float));
  float* hB      = (float*)(ws + off); off += align256((size_t)N * nhid * sizeof(float));
  float* w1T     = (float*)(ws + off); off += align256((size_t)nfeat * nhid * sizeof(float));
  float* wxT     = (float*)(ws + off); off += align256((size_t)nhid * nhid * sizeof(float));
  float* w2T     = (float*)(ws + off);

  build_row_ptr<<<(E + 255) / 256, 256, 0, stream>>>(erow, E, N, rp);
  transpose_mat<<<(nfeat * nhid + 255) / 256, 256, 0, stream>>>(W1, w1T, nfeat, nhid);
  transpose_mat<<<(nhid * nhid + 255) / 256, 256, 0, stream>>>(Wx, wxT, nhid, nhid);
  transpose_mat<<<(nhid * nclass + 255) / 256, 256, 0, stream>>>(W2, w2T, nhid, nclass);

  // Layer 1: h = relu(spmm(x @ W1) + b1)
  launch_gemm(x, w1T, support, N, nfeat, nhid, stream);
  spmm_bias_act<<<N, 64, 0, stream>>>(rp, ecol, eval_, support, b1, hA, nhid, 1);

  // 8 shared-weight hidden layers
  float* cur = hA;
  float* nxt = hB;
  for (int l = 0; l < 8; ++l) {
    launch_gemm(cur, wxT, support, N, nhid, nhid, stream);
    spmm_bias_act<<<N, 64, 0, stream>>>(rp, ecol, eval_, support, bx, nxt, nhid, 1);
    float* t = cur; cur = nxt; nxt = t;
  }

  // Output layer: logits = spmm(h @ W2) + b2, then log_softmax
  launch_gemm(cur, w2T, support, N, nhid, nclass, stream);
  spmm_bias_act<<<N, 64, 0, stream>>>(rp, ecol, eval_, support, b2, (float*)d_out, nclass, 0);
  log_softmax_rows<<<(N + 7) / 8, 256, 0, stream>>>((float*)d_out, N, nclass);
}
